// TorchMoE_54185307406692
// MI455X (gfx1250) — matrix operand order repaired and checked by reading, not compiled, not run on hardware
//
#include <hip/hip_runtime.h>
#include <hip/hip_bf16.h>

// ---------------- problem constants (from reference) ----------------
#define D_     1024
#define H_     2048
#define TWOH_  4096
#define NEXP_  8
#define NROUT_ 7
#define TILE_M 32
#define XS_STRIDE (D_ + 8)   // 1032 bf16: 516 dwords %64 banks = 4 -> skewed
#define HS_STRIDE (H_ + 8)   // 2056 bf16: 1028 dwords %64 banks = 4 -> skewed
#define PK_COLS  512         // packing stage width (bf16 cols per LDS tile)
#define PK_LDST  (PK_COLS + 8)

#define AS1 __attribute__((address_space(1)))
#define AS3 __attribute__((address_space(3)))

#if __has_builtin(__builtin_amdgcn_global_load_async_to_lds_b128)
#define HAVE_ASYNC_LDS 1
#else
#define HAVE_ASYNC_LDS 0
#endif

typedef __attribute__((ext_vector_type(16))) __bf16 v16bf;
typedef __attribute__((ext_vector_type(8)))  __bf16 v8bf;
typedef __attribute__((ext_vector_type(8)))  float  v8f;
typedef int v4i __attribute__((vector_size(16)));   // matches builtin's V4i
typedef AS1 v4i* gv4p;                              // global-AS pointer to b128
typedef AS3 v4i* lv4p;                              // LDS-AS pointer to b128

static __device__ inline v16bf make_v16(uint4 lo, uint4 hi) {
    union { uint4 u; v8bf v; } a, b;
    a.u = lo; b.u = hi;
    return __builtin_shufflevector(a.v, b.v,
        0, 1, 2, 3, 4, 5, 6, 7, 8, 9, 10, 11, 12, 13, 14, 15);
}

static __device__ inline v8f wmma_bf16(v16bf a, v16bf b, v8f c) {
    return __builtin_amdgcn_wmma_f32_16x16x32_bf16(false, a, false, b,
                                                   (short)0, c, false, false);
}

static __device__ inline float silu_mul(float a, float b) {
    return a * (b / (1.0f + __expf(-b)));
}

// ---------------- fp32 -> bf16 conversion (x activations, vectorized x4) ----------------
__global__ __launch_bounds__(256)
void cvt_kernel(const float* __restrict__ s, __bf16* __restrict__ d, size_t n4) {
    size_t i = (size_t)blockIdx.x * blockDim.x + threadIdx.x;
    size_t stride = (size_t)gridDim.x * blockDim.x;
    const float4* s4 = reinterpret_cast<const float4*>(s);
    for (; i < n4; i += stride) {
        float4 v = s4[i];
        union { __bf16 h[4]; unsigned long long u; } p;
        p.h[0] = (__bf16)v.x; p.h[1] = (__bf16)v.y;
        p.h[2] = (__bf16)v.z; p.h[3] = (__bf16)v.w;
        *reinterpret_cast<unsigned long long*>(d + i * 4) = p.u;
    }
}

// ---------------- weight pack: fp32 [K x N] row-major -> bf16 WMMA tiles ----------------
// tile(nt, kb) = contiguous 1KB block of 32(K) x 16(N) bf16; lane l slot s (nn=l%16, kk table order) at
// ((nt*KB + kb)*32 + l)*16 + s.  B-fragment loads become lane-contiguous 1KB blocks.
__global__ __launch_bounds__(256)
void pack_kernel(const float* __restrict__ src, __bf16* __restrict__ dst,
                 int K, int Ncols) {
    __shared__ __bf16 ls[32 * PK_LDST];      // 33.3 KB
    const int cc = blockIdx.x, kb = blockIdx.y, e = blockIdx.z;
    const int tid = threadIdx.x;
    const size_t esz = (size_t)K * Ncols;
    const float* S = src + (size_t)e * esz;
    __bf16*      Dt = dst + (size_t)e * esz;
    const int KB = K / 32;

    // stage + convert: rows kb*32..+31, cols cc*512..+511 (coalesced row segments)
    for (int i = tid; i < 32 * (PK_COLS / 4); i += 256) {   // 4096 float4
        int r  = i >> 7;         // /128
        int c4 = i & 127;
        float4 v = reinterpret_cast<const float4*>(
            S + (size_t)(kb * 32 + r) * Ncols + cc * PK_COLS)[c4];
        union { __bf16 h[4]; unsigned long long u; } p;
        p.h[0] = (__bf16)v.x; p.h[1] = (__bf16)v.y;
        p.h[2] = (__bf16)v.z; p.h[3] = (__bf16)v.w;
        *reinterpret_cast<unsigned long long*>(&ls[r * PK_LDST + c4 * 4]) = p.u;
    }
    __syncthreads();

    // write 32 tiles; each wave writes one fully-contiguous 1KB tile per iteration
    const int lane = tid & 31, w = tid >> 5;
    for (int j = 0; j < 4; ++j) {
        int ntl = w + 8 * j;
        int nt  = cc * (PK_COLS / 16) + ntl;
        union { __bf16 h[16]; uint4 u[2]; } g;
        for (int s = 0; s < 16; ++s) {
            int kk = (s < 8) ? 8 * (lane >> 4) + s : 16 + 8 * (lane >> 4) + (s - 8);
            g.h[s] = ls[kk * PK_LDST + ntl * 16 + (lane & 15)];
        }
        uint4 lo = g.u[0], hi = g.u[1];
        uint4* dp = reinterpret_cast<uint4*>(Dt + ((size_t)(nt * KB + kb) * 32 + lane) * 16);
        dp[0] = lo;
        dp[1] = hi;
    }
}

// ---------------- zero helpers ----------------
__global__ void zero_counts_kernel(int* __restrict__ counts) {
    if (threadIdx.x < NEXP_) counts[threadIdx.x] = 0;
}

__global__ __launch_bounds__(256)
void zero_y_kernel(float4* __restrict__ y, size_t n4) {
    size_t i = (size_t)blockIdx.x * blockDim.x + threadIdx.x;
    size_t stride = (size_t)gridDim.x * blockDim.x;
    float4 z = {0.f, 0.f, 0.f, 0.f};
    for (; i < n4; i += stride) y[i] = z;
}

// ---------------- gating: argmax over routed logits, bucket tokens ----------------
__global__ __launch_bounds__(256)
void gate_kernel(const float* __restrict__ x, const float* __restrict__ Wg,
                 int* __restrict__ counts, int* __restrict__ lists, int M) {
    __shared__ float red[NROUT_][256];
    int m = blockIdx.x, tid = threadIdx.x;
    float s[NROUT_];
#pragma unroll
    for (int n = 0; n < NROUT_; ++n) s[n] = 0.f;
    const float* xr = x + (size_t)m * D_;
    for (int d = tid; d < D_; d += 256) {
        float xv = xr[d];
        const float* wr = Wg + (size_t)d * NEXP_;
#pragma unroll
        for (int n = 0; n < NROUT_; ++n) s[n] += xv * wr[n];
    }
#pragma unroll
    for (int n = 0; n < NROUT_; ++n) red[n][tid] = s[n];
    __syncthreads();
    for (int st = 128; st > 0; st >>= 1) {
        if (tid < st)
#pragma unroll
            for (int n = 0; n < NROUT_; ++n) red[n][tid] += red[n][tid + st];
        __syncthreads();
    }
    if (tid == 0) {
        int best = 0; float bv = red[0][0];
#pragma unroll
        for (int n = 1; n < NROUT_; ++n) {
            float v = red[n][0];
            if (v > bv) { bv = v; best = n; }  // first index wins ties, like top_k
        }
        int pos = atomicAdd(&counts[best], 1);
        lists[best * M + pos] = m;
    }
}

// ---------------- fused SwiGLU expert: z = xW1, h = silu(b)*a, y += hW2 ----------------
// One workgroup = 32-token tile of one expert. 512 threads = 16 waves, 1 WG/WGP.
// All 8 experts run concurrently; each y element gets exactly 2 atomic adds
// (shared + its routed expert), which is bitwise order-invariant.
__global__ __launch_bounds__(512)
void expert_kernel(const __bf16* __restrict__ xbf,
                   const __bf16* __restrict__ W1p,  // [8] packed tiles: nt<256, KB=32
                   const __bf16* __restrict__ W2p,  // [8] packed tiles: nt<64,  KB=64
                   float* __restrict__ y,           // [M][D]
                   const int* __restrict__ counts,
                   const int* __restrict__ lists,   // [7][M]
                   int M) {
    __shared__ __bf16 xs[TILE_M * XS_STRIDE];   //  66.0 KB
    __shared__ __bf16 hs[TILE_M * HS_STRIDE];   // 131.6 KB
    __shared__ int tokv[TILE_M];
    __shared__ int rows_s;

    const int tile   = blockIdx.x;
    const int e      = blockIdx.y;           // 0..6 routed, 7 shared
    const int routed = (e != NEXP_ - 1);
    const int tid    = threadIdx.x;
    const int lane   = tid & 31;
    const int w      = tid >> 5;             // wave id 0..15

    const int base = tile * TILE_M;
    int rows = TILE_M;
    if (routed) {
        int cnt = counts[e];
        if (base >= cnt) return;             // uniform exit: EXEC all-1s for WMMA
        rows = min(TILE_M, cnt - base);
    }
    if (tid < TILE_M) {
        int r = tid, tok;
        if (routed) tok = (r < rows) ? lists[e * M + base + r] : lists[e * M + base];
        else        tok = base + r;
        tokv[r] = tok;
    }
    if (tid == 0) rows_s = rows;
    __syncthreads();

    // stage x tile (32 rows x 1024 bf16 = 128 uint4/row) into LDS.
    // CDNA5 path: async global->LDS copies (ASYNCcnt), bypassing VGPRs.
    for (int i = tid; i < TILE_M * (D_ / 8); i += 512) {
        int r = i >> 7, c = i & 127;
        const __bf16* gsrc = xbf + (size_t)tokv[r] * D_ + c * 8;
        __bf16* ldst = &xs[r * XS_STRIDE + c * 8];
#if HAVE_ASYNC_LDS
        __builtin_amdgcn_global_load_async_to_lds_b128(
            (gv4p)(unsigned long long)gsrc,
            (lv4p)(unsigned)(unsigned long long)ldst,
            0, 0);
#else
        *reinterpret_cast<uint4*>(ldst) = *reinterpret_cast<const uint4*>(gsrc);
#endif
    }
#if HAVE_ASYNC_LDS
#if __has_builtin(__builtin_amdgcn_s_wait_asynccnt)
    __builtin_amdgcn_s_wait_asynccnt(0);
#else
    asm volatile("s_wait_asynccnt 0" ::: "memory");
#endif
#endif
    __syncthreads();

    const __bf16* W1e = W1p + (size_t)e * D_ * TWOH_;
    const __bf16* W2e = W2p + (size_t)e * H_ * D_;
    const int arow = lane & 15;
    const int asel = (lane >> 4) << 3;  // lanes 0-15: K{0..7,16..23}; 16-31: K{8..15,24..31}

    // ---- phase A: z = x @ W1e. 2 row-tiles x (a, b=a+H) share A/B fragments ----
    for (int t = 0; t < 8; ++t) {
        const int ntA = w + 16 * t;          // 'a' n-tile   [0, 128)
        const int ntB = ntA + 128;           // 'b' n-tile   [128, 256)
        // packed streams: 32 consecutive 1KB tiles each, lane-contiguous
        const __bf16* pA = W1e + (size_t)ntA * (D_ / 32) * 512 + lane * 16;
        const __bf16* pB = W1e + (size_t)ntB * (D_ / 32) * 512 + lane * 16;
        if (t < 7) {            // pull next panel's streams toward the WGP
            __builtin_prefetch(pA + (size_t)16 * (D_ / 32) * 512, 0, 1);
            __builtin_prefetch(pB + (size_t)16 * (D_ / 32) * 512, 0, 1);
        }
        v8f acc00 = {}, acc01 = {}, acc10 = {}, acc11 = {};
        for (int kb = 0; kb < D_ / 32; ++kb) {
            const int k0 = kb * 32 + asel;
            const uint4* ap0 = reinterpret_cast<const uint4*>(&xs[arow * XS_STRIDE + k0]);
            const uint4* ap1 = reinterpret_cast<const uint4*>(&xs[(arow + 16) * XS_STRIDE + k0]);
            v16bf a0 = make_v16(ap0[0], ap0[2]);       // +16 elems = +2 uint4
            v16bf a1 = make_v16(ap1[0], ap1[2]);
            const uint4* bpA = reinterpret_cast<const uint4*>(pA + kb * 512);
            const uint4* bpB = reinterpret_cast<const uint4*>(pB + kb * 512);
            v16bf bA = make_v16(bpA[0], bpA[1]);
            v16bf bB = make_v16(bpB[0], bpB[1]);
            acc00 = wmma_bf16(a0, bA, acc00);
            acc01 = wmma_bf16(a0, bB, acc01);
            acc10 = wmma_bf16(a1, bA, acc10);
            acc11 = wmma_bf16(a1, bB, acc11);
        }
        // SwiGLU in-register, spill h tile (bf16) to LDS
        const int col = ntA * 16 + (lane & 15);
#pragma unroll
        for (int i = 0; i < 8; ++i) {
            int r = (lane < 16) ? i : i + 8;
            hs[r * HS_STRIDE + col]        = (__bf16)silu_mul(acc00[i], acc01[i]);
            hs[(r + 16) * HS_STRIDE + col] = (__bf16)silu_mul(acc10[i], acc11[i]);
        }
    }
    __syncthreads();

    // ---- phase B: y += h @ W2e. 2 row-tiles x 2 n-tiles share fragments ----
    for (int u = 0; u < 2; ++u) {
        const int nt0 = w + 16 * u;                // [0, 32)
        const int nt1 = nt0 + 32;                  // [32, 64)
        const __bf16* p0 = W2e + (size_t)nt0 * (H_ / 32) * 512 + lane * 16;
        const __bf16* p1 = W2e + (size_t)nt1 * (H_ / 32) * 512 + lane * 16;
        if (u == 0) {
            __builtin_prefetch(p0 + (size_t)16 * (H_ / 32) * 512, 0, 1);
            __builtin_prefetch(p1 + (size_t)16 * (H_ / 32) * 512, 0, 1);
        }
        v8f acc00 = {}, acc01 = {}, acc10 = {}, acc11 = {};
        for (int kb = 0; kb < H_ / 32; ++kb) {
            const int k0 = kb * 32 + asel;
            const uint4* ap0 = reinterpret_cast<const uint4*>(&hs[arow * HS_STRIDE + k0]);
            const uint4* ap1 = reinterpret_cast<const uint4*>(&hs[(arow + 16) * HS_STRIDE + k0]);
            v16bf a0 = make_v16(ap0[0], ap0[2]);
            v16bf a1 = make_v16(ap1[0], ap1[2]);
            const uint4* bp0 = reinterpret_cast<const uint4*>(p0 + kb * 512);
            const uint4* bp1 = reinterpret_cast<const uint4*>(p1 + kb * 512);
            v16bf b0 = make_v16(bp0[0], bp0[1]);
            v16bf b1 = make_v16(bp1[0], bp1[1]);
            acc00 = wmma_bf16(a0, b0, acc00);
            acc01 = wmma_bf16(a0, b1, acc01);
            acc10 = wmma_bf16(a1, b0, acc10);
            acc11 = wmma_bf16(a1, b1, acc11);
        }
        const int c0 = nt0 * 16 + (lane & 15);
        const int c1 = nt1 * 16 + (lane & 15);
#pragma unroll
        for (int i = 0; i < 8; ++i) {
            int r = (lane < 16) ? i : i + 8;
            if (r < rows_s) {
                size_t ro = (size_t)tokv[r] * D_;
                atomicAdd(&y[ro + c0], acc00[i]);      // exactly 2 adders per element:
                atomicAdd(&y[ro + c1], acc01[i]);      // order-invariant bitwise
            }
            if (r + 16 < rows_s) {
                size_t ro = (size_t)tokv[r + 16] * D_;
                atomicAdd(&y[ro + c0], acc10[i]);
                atomicAdd(&y[ro + c1], acc11[i]);
            }
        }
    }
}

// ---------------- host-side orchestration ----------------
static inline size_t align256(size_t x) { return (x + 255) & ~(size_t)255; }

extern "C" void kernel_launch(void* const* d_in, const int* in_sizes, int n_in,
                              void* d_out, int out_size, void* d_ws, size_t ws_size,
                              hipStream_t stream) {
    const float* x  = (const float*)d_in[0];   // [B,S,D]
    const float* Wg = (const float*)d_in[1];   // [D,N]
    const float* W1 = (const float*)d_in[2];   // [N,D,2H]
    const float* W2 = (const float*)d_in[3];   // [N,H,D]
    float* y = (float*)d_out;                  // [M,D]

    const int M = in_sizes[0] / D_;            // 2048 tokens
    const size_t nx = (size_t)M * D_;
    const size_t n1 = (size_t)NEXP_ * D_ * TWOH_;
    const size_t n2 = (size_t)NEXP_ * H_ * D_;

    char* ws = (char*)d_ws;
    __bf16* xbf = (__bf16*)ws;            ws += align256(nx * sizeof(__bf16));
    __bf16* W1b = (__bf16*)ws;            ws += align256(n1 * sizeof(__bf16));
    __bf16* W2b = (__bf16*)ws;            ws += align256(n2 * sizeof(__bf16));
    int* counts = (int*)ws;               ws += align256(32 * sizeof(int));
    int* lists  = (int*)ws;               /* [7][M] ints */

    // 1) precision + layout pass: x -> bf16, weights -> bf16 packed WMMA tiles
    //    (96MB bf16 weight set then lives in the 192MB L2)
    cvt_kernel<<<512, 256, 0, stream>>>(x, xbf, nx / 4);
    pack_kernel<<<dim3(TWOH_ / PK_COLS, D_ / 32, NEXP_), 256, 0, stream>>>(W1, W1b, D_, TWOH_);
    pack_kernel<<<dim3(D_ / PK_COLS,    H_ / 32, NEXP_), 256, 0, stream>>>(W2, W2b, H_, D_);

    // 2) routing + output zero-init (y receives pure atomic accumulation)
    zero_counts_kernel<<<1, 32, 0, stream>>>(counts);
    zero_y_kernel<<<512, 256, 0, stream>>>((float4*)y, nx / 4);
    gate_kernel<<<M, 256, 0, stream>>>(x, Wg, counts, lists, M);

    // 3) all experts concurrently (e=0..6 routed buckets, e=7 shared over all tokens)
    expert_kernel<<<dim3(M / TILE_M, NEXP_), 512, 0, stream>>>(
        xbf, W1b, W2b, y, counts, lists, M);
}
